// MultiHeadedAttention_29557964931669
// MI455X (gfx1250) — matrix operand order repaired and checked by reading, not compiled, not run on hardware
//
#include <hip/hip_runtime.h>
#include <hip/hip_bf16.h>

constexpr int kB  = 2;
constexpr int kS  = 2048;
constexpr int kD  = 1024;
constexpr int kH  = 16;
constexpr int kDk = 64;

typedef __attribute__((ext_vector_type(16))) __bf16 v16bf;
typedef __attribute__((ext_vector_type(8)))  float  v8f;
typedef __attribute__((ext_vector_type(4)))  unsigned v4u;
typedef __attribute__((ext_vector_type(8)))  unsigned v8u;

union Frag {
  unsigned u[8];
  v16bf    v;
};

static __device__ __forceinline__ unsigned short f2bf(float f) {
  unsigned u;
  __builtin_memcpy(&u, &f, 4);
  u += 0x7FFFu + ((u >> 16) & 1u);   // round-to-nearest-even
  return (unsigned short)(u >> 16);
}
static __device__ __forceinline__ float bf2f(unsigned short h) {
  unsigned u = ((unsigned)h) << 16;
  float f;
  __builtin_memcpy(&f, &u, 4);
  return f;
}
static __device__ __forceinline__ unsigned pack2(float lo, float hi) {
  return (unsigned)f2bf(lo) | ((unsigned)f2bf(hi) << 16);
}

static __device__ __forceinline__ v8f wmma_bf16(const Frag& a, const Frag& b, v8f c) {
  return __builtin_amdgcn_wmma_f32_16x16x32_bf16(
      false, a.v, false, b.v, (short)0, c, false, false);
}

// ---- Tensor Data Mover: async 2D bf16 tile load, global -> LDS --------------
// D# per CDNA5 ISA. All operands uniform => SGPR groups.
static __device__ __forceinline__ void tdm_load_2d(
    unsigned lds_off, const void* gptr,
    unsigned tensor_d0, unsigned tensor_d1, unsigned long long stride0_elems,
    unsigned tile_d0, unsigned tile_d1) {
  unsigned long long ga = (unsigned long long)gptr;
  v4u g0;
  g0[0] = 1u;                                                // count=1, user mode
  g0[1] = lds_off;                                           // lds_addr (bytes)
  g0[2] = (unsigned)ga;                                      // global_addr[31:0]
  g0[3] = (unsigned)((ga >> 32) & 0x01FFFFFFu) | (2u << 30); // addr[56:32]|type=2
  v8u g1;
  g1[0] = 0x10000u;                                 // wg_mask=0, data_size=1 (2B)
  g1[1] = (tensor_d0 & 0xFFFFu) << 16;              // tensor_dim0[15:0]
  g1[2] = (tensor_d0 >> 16) | ((tensor_d1 & 0xFFFFu) << 16);
  g1[3] = (tensor_d1 >> 16) | (tile_d0 << 16);      // tile_dim0
  g1[4] = tile_d1;                                  // tile_dim1 (tile_dim2=0)
  g1[5] = (unsigned)stride0_elems;                  // tensor_dim0_stride[31:0]
  g1[6] = (unsigned)(stride0_elems >> 32) & 0xFFFFu;
  g1[7] = 0u;
  v4u gz = {0u, 0u, 0u, 0u};                        // groups 2/3: dims unused
  asm volatile("tensor_load_to_lds %0, %1, %2, %3"
               :: "s"(g0), "s"(g1), "s"(gz), "s"(gz)
               : "memory");
}

// ---- Fragment loaders (wave32 layouts per CDNA5 ISA 7.12.2) -----------------
static __device__ __forceinline__ Frag loadA_bf(const unsigned short* p, int ld,
                                                int row0, int k0, int rowmax) {
  Frag f;
  int lane = threadIdx.x & 31;
  int row = row0 + (lane & 15);
  if (row > rowmax) row = rowmax;
  int kb = (lane < 16) ? 0 : 8;
  const unsigned short* base = p + (size_t)row * ld + k0;
#pragma unroll
  for (int t = 0; t < 8; ++t) {
    int K = kb + ((t < 4) ? (2 * t) : (8 + 2 * t));
    f.u[t] = *(const unsigned*)(base + K);
  }
  return f;
}

static __device__ __forceinline__ Frag loadA_f32(const float* p, int ld,
                                                 int row0, int k0, int rowmax) {
  Frag f;
  int lane = threadIdx.x & 31;
  int row = row0 + (lane & 15);
  if (row > rowmax) row = rowmax;
  int kb = (lane < 16) ? 0 : 8;
  const float* base = p + (size_t)row * ld + k0;
#pragma unroll
  for (int t = 0; t < 8; ++t) {
    int K = kb + ((t < 4) ? (2 * t) : (8 + 2 * t));
    f.u[t] = pack2(base[K], base[K + 1]);
  }
  return f;
}

static __device__ __forceinline__ Frag loadB_bf(const unsigned short* p, int ld,
                                                int n0, int k0) {
  Frag f;
  int lane = threadIdx.x & 31;
  int n = n0 + (lane & 15);
  int kb = (lane < 16) ? 0 : 8;
  const unsigned short* base = p + (size_t)n * ld + k0 + kb;
#pragma unroll
  for (int t = 0; t < 8; ++t)
    f.u[t] = *(const unsigned*)(base + ((t < 4) ? (2 * t) : (8 + 2 * t)));
  return f;
}

// ---- Kernel 1: fp32 -> bf16 conversion (weights) ----------------------------
__global__ void cvt_bf16_kernel(const float* __restrict__ src,
                                unsigned short* __restrict__ dst, int n) {
  int i = blockIdx.x * blockDim.x + threadIdx.x;
  if (i < n) dst[i] = f2bf(src[i]);
}

// ---- Kernel 2: WMMA GEMM, 32x64 C-tile per wave -----------------------------
// ABF=false: A operand fp32 (converted in-loader); ABF=true: A operand bf16.
// mode 0: K-proj -> (b,h,s,d) bf16 +bias   | mode 1: Q-proj -> two outs +rwb/+rb
// mode 2: V-proj -> (b,h,d,s) bf16 +bias   | mode 3: R-proj -> (h,s,d) bf16
// mode 4: O-proj -> fp32 (m,n) +bias
template <bool ABF>
__global__ __launch_bounds__(32) void proj_kernel(
    const float* __restrict__ Xf, const unsigned short* __restrict__ Abf,
    const unsigned short* __restrict__ Wb, const float* __restrict__ bias,
    const float* __restrict__ rwb, const float* __restrict__ rb,
    unsigned short* __restrict__ out0, unsigned short* __restrict__ out1,
    float* __restrict__ outf, int M, int mode) {
  int n0 = blockIdx.x * 64;
  int m0 = blockIdx.y * 32;
  int lane = threadIdx.x & 31;

  v8f c[2][4];
#pragma unroll
  for (int mi = 0; mi < 2; ++mi)
#pragma unroll
    for (int i = 0; i < 4; ++i)
#pragma unroll
      for (int e = 0; e < 8; ++e) c[mi][i][e] = 0.f;

  for (int k0 = 0; k0 < kD; k0 += 32) {
    Frag a0, a1;
    if constexpr (ABF) {
      a0 = loadA_bf(Abf, kD, m0, k0, M - 1);
      a1 = loadA_bf(Abf, kD, m0 + 16, k0, M - 1);
    } else {
      a0 = loadA_f32(Xf, kD, m0, k0, M - 1);
      a1 = loadA_f32(Xf, kD, m0 + 16, k0, M - 1);
    }
#pragma unroll
    for (int ns = 0; ns < 4; ++ns) {
      Frag b = loadB_bf(Wb, kD, n0 + ns * 16, k0);
      c[0][ns] = wmma_bf16(a0, b, c[0][ns]);
      c[1][ns] = wmma_bf16(a1, b, c[1][ns]);
    }
  }

  int hi = lane >> 4, col = lane & 15;
#pragma unroll
  for (int mi = 0; mi < 2; ++mi) {
#pragma unroll
    for (int ns = 0; ns < 4; ++ns) {
#pragma unroll
      for (int t = 0; t < 8; ++t) {
        int m = m0 + mi * 16 + t + 8 * hi;
        int n = n0 + ns * 16 + col;
        float v = c[mi][ns][t];
        if (bias) v += bias[n];
        int h = n >> 6, d = n & 63;
        if (mode == 4) {
          outf[(size_t)m * kD + n] = v;
        } else if (mode == 3) {
          out0[((size_t)h * kS + m) * kDk + d] = f2bf(v);
        } else {
          int b = m / kS, s = m % kS;
          if (mode == 0) {
            out0[(((size_t)(b * kH + h) * kS) + s) * kDk + d] = f2bf(v);
          } else if (mode == 1) {
            size_t o = (((size_t)(b * kH + h) * kS) + s) * kDk + d;
            out0[o] = f2bf(v + rwb[n]);
            out1[o] = f2bf(v + rb[n]);
          } else {
            out0[(((size_t)(b * kH + h) * kDk) + d) * kS + s] = f2bf(v);
          }
        }
      }
    }
  }
}

// ---- Kernel 3: fused XL attention, TDM double-buffered K/V ------------------
// LDS map (bytes): sBD [0,69632)  sP [69632,70656)
//                  sK  [70656, +2*4096)  sV [78848, +2*4096)   total 87040
__global__ __launch_bounds__(32) void attn_kernel(
    const unsigned short* __restrict__ qw_, const unsigned short* __restrict__ qr_,
    const unsigned short* __restrict__ kk_, const unsigned short* __restrict__ vT_,
    const unsigned short* __restrict__ rr_, unsigned short* __restrict__ obf) {
  extern __shared__ char smem[];
  unsigned short* sBD = (unsigned short*)smem;    // 17 x kS bf16
  unsigned short* sP  = sBD + 17 * kS;            // 16 x 32 bf16
  unsigned short* sK  = (unsigned short*)(smem + 70656);  // 2 x [32][64]
  unsigned short* sV  = (unsigned short*)(smem + 78848);  // 2 x [64][32]
  const unsigned ldsK = 70656u, ldsV = 78848u;

  int i0 = blockIdx.x * 16;
  int h  = blockIdx.y;
  int b  = blockIdx.z;
  int lane = threadIdx.x & 31;
  int hi = lane >> 4, col = lane & 15;

  const unsigned short* qw = qw_ + ((size_t)(b * kH + h) * kS) * kDk;
  const unsigned short* qr = qr_ + ((size_t)(b * kH + h) * kS) * kDk;
  const unsigned short* kk = kk_ + ((size_t)(b * kH + h) * kS) * kDk;
  const unsigned short* vT = vT_ + ((size_t)(b * kH + h) * kDk) * kS;
  const unsigned short* rr = rr_ + ((size_t)h * kS) * kDk;

  Frag qwA[2], qrA[2], qrA2[2];
#pragma unroll
  for (int kc = 0; kc < 2; ++kc) {
    qwA[kc]  = loadA_bf(qw, kDk, i0, kc * 32, kS - 1);
    qrA[kc]  = loadA_bf(qr, kDk, i0, kc * 32, kS - 1);
    qrA2[kc] = loadA_bf(qr, kDk, i0 + 16, kc * 32, kS - 1);
  }

  // ---- Phase 1: BD[i0..i0+16][0..S) via WMMA into LDS ----
  for (int jt = 0; jt < kS / 16; ++jt) {
    int j0 = jt * 16;
    v8f c0, c1;
#pragma unroll
    for (int e = 0; e < 8; ++e) { c0[e] = 0.f; c1[e] = 0.f; }
#pragma unroll
    for (int kc = 0; kc < 2; ++kc) {
      Frag bR = loadB_bf(rr, kDk, j0, kc * 32);
      c0 = wmma_bf16(qrA[kc], bR, c0);
      c1 = wmma_bf16(qrA2[kc], bR, c1);
    }
#pragma unroll
    for (int t = 0; t < 8; ++t)
      sBD[(t + 8 * hi) * kS + j0 + col] = f2bf(c0[t]);
    if (hi == 0) sBD[16 * kS + j0 + col] = f2bf(c1[0]);
  }

  // ---- Phase 2: streaming softmax; K/V tiles via async TDM double buffer ----
  float mrow[8], lrow[8];
  v8f acc[4];
#pragma unroll
  for (int t = 0; t < 8; ++t) { mrow[t] = -1e30f; lrow[t] = 0.f; }
#pragma unroll
  for (int d4 = 0; d4 < 4; ++d4)
#pragma unroll
    for (int e = 0; e < 8; ++e) acc[d4][e] = 0.f;

  // prologue: tile 0 into buffer 0  (TENSORcnt += 2)
  tdm_load_2d(ldsK, kk, kDk, kS, kDk, kDk, 32);
  tdm_load_2d(ldsV, vT, kS, kDk, kS, 32, kDk);

  for (int jt = 0; jt < kS / 32; ++jt) {
    int j0 = jt * 32;
    int buf = jt & 1;
    if (jt + 1 < kS / 32) {
      int j1 = j0 + 32;
      tdm_load_2d(ldsK + (buf ^ 1) * 4096u, kk + (size_t)j1 * kDk,
                  kDk, kS, kDk, kDk, 32);
      tdm_load_2d(ldsV + (buf ^ 1) * 4096u, vT + j1, kS, kDk, kS, 32, kDk);
      __builtin_amdgcn_s_wait_tensorcnt((unsigned short)2);  // cur buf landed
    } else {
      __builtin_amdgcn_s_wait_tensorcnt((unsigned short)0);
    }
    const unsigned short* kT = sK + buf * 2048;  // [32 keys][64 ch]
    const unsigned short* vL = sV + buf * 2048;  // [64 d][32 keys]

    v8f ac[2];
#pragma unroll
    for (int ns = 0; ns < 2; ++ns) {
#pragma unroll
      for (int e = 0; e < 8; ++e) ac[ns][e] = 0.f;
#pragma unroll
      for (int kc = 0; kc < 2; ++kc)
        ac[ns] = wmma_bf16(qwA[kc], loadB_bf(kT, kDk, ns * 16, kc * 32), ac[ns]);
    }
    // rel-shift: f=(i+1)S+j; i'=f/(S+1); m=f%(S+1); bd=(m==0)?0:BD[i'][m-1]
    float p[2][8], tmax[8];
#pragma unroll
    for (int t = 0; t < 8; ++t) tmax[t] = -1e30f;
#pragma unroll
    for (int ns = 0; ns < 2; ++ns)
#pragma unroll
      for (int t = 0; t < 8; ++t) {
        int i = i0 + t + 8 * hi;
        int j = j0 + ns * 16 + col;
        unsigned f = (unsigned)(i + 1) * kS + (unsigned)j;
        unsigned i2 = f / (kS + 1);
        unsigned m  = f - i2 * (kS + 1);
        float bd = (m == 0u) ? 0.f
                             : bf2f(sBD[((int)i2 - i0) * kS + ((int)m - 1)]);
        float val = (ac[ns][t] + bd) * 0.125f;
        p[ns][t] = val;
        tmax[t] = fmaxf(tmax[t], val);
      }
#pragma unroll
    for (int mask = 1; mask < 16; mask <<= 1)
#pragma unroll
      for (int t = 0; t < 8; ++t)
        tmax[t] = fmaxf(tmax[t], __shfl_xor(tmax[t], mask, 32));
    float alpha[8], rsum[8];
#pragma unroll
    for (int t = 0; t < 8; ++t) {
      float mn = fmaxf(mrow[t], tmax[t]);
      alpha[t] = __expf(mrow[t] - mn);
      mrow[t] = mn;
      rsum[t] = 0.f;
    }
#pragma unroll
    for (int ns = 0; ns < 2; ++ns)
#pragma unroll
      for (int t = 0; t < 8; ++t) {
        float e = __expf(p[ns][t] - mrow[t]);
        p[ns][t] = e;
        rsum[t] += e;
      }
#pragma unroll
    for (int mask = 1; mask < 16; mask <<= 1)
#pragma unroll
      for (int t = 0; t < 8; ++t) rsum[t] += __shfl_xor(rsum[t], mask, 32);
#pragma unroll
    for (int t = 0; t < 8; ++t) lrow[t] = lrow[t] * alpha[t] + rsum[t];
#pragma unroll
    for (int d4 = 0; d4 < 4; ++d4)
#pragma unroll
      for (int t = 0; t < 8; ++t) acc[d4][t] *= alpha[t];
    // P (C-frag) -> A-frag via LDS, then O += P * V
#pragma unroll
    for (int ns = 0; ns < 2; ++ns)
#pragma unroll
      for (int t = 0; t < 8; ++t)
        sP[(t + 8 * hi) * 32 + ns * 16 + col] = f2bf(p[ns][t]);
    Frag pA = loadA_bf(sP, 32, 0, 0, 15);
#pragma unroll
    for (int d4 = 0; d4 < 4; ++d4)
      acc[d4] = wmma_bf16(pA, loadB_bf(vL, 32, d4 * 16, 0), acc[d4]);
  }

#pragma unroll
  for (int d4 = 0; d4 < 4; ++d4)
#pragma unroll
    for (int t = 0; t < 8; ++t) {
      int i = i0 + t + 8 * hi;
      int d = d4 * 16 + col;
      float v = acc[d4][t] / lrow[t];
      obf[((size_t)(b * kS + i)) * kD + h * kDk + d] = f2bf(v);
    }
}

extern "C" void kernel_launch(void* const* d_in, const int* in_sizes, int n_in,
                              void* d_out, int out_size, void* d_ws, size_t ws_size,
                              hipStream_t stream) {
  (void)in_sizes; (void)n_in; (void)out_size; (void)ws_size;
  const float* query    = (const float*)d_in[0];
  const float* key_     = (const float*)d_in[1];
  const float* value    = (const float*)d_in[2];
  const float* r_emb    = (const float*)d_in[3];
  const float* r_bias   = (const float*)d_in[4];
  const float* r_w_bias = (const float*)d_in[5];
  const float* Wq = (const float*)d_in[6];
  const float* bq = (const float*)d_in[7];
  const float* Wk = (const float*)d_in[8];
  const float* bk = (const float*)d_in[9];
  const float* Wv = (const float*)d_in[10];
  const float* bv = (const float*)d_in[11];
  const float* Wr = (const float*)d_in[12];
  const float* Wo = (const float*)d_in[13];
  const float* bo = (const float*)d_in[14];
  float* out = (float*)d_out;

  unsigned short* w = (unsigned short*)d_ws;
  const size_t nW   = (size_t)kD * kD;
  const size_t nAct = (size_t)kB * kH * kS * kDk;
  unsigned short* Wqb = w; w += nW;
  unsigned short* Wkb = w; w += nW;
  unsigned short* Wvb = w; w += nW;
  unsigned short* Wrb = w; w += nW;
  unsigned short* Wob = w; w += nW;
  unsigned short* qwb = w; w += nAct;   // (b,h,s,d)  proj_q + r_w_bias
  unsigned short* qrb = w; w += nAct;   // (b,h,s,d)  proj_q + r_bias
  unsigned short* kbf = w; w += nAct;   // (b,h,s,d)
  unsigned short* vtb = w; w += nAct;   // (b,h,d,s)
  unsigned short* rrb = w; w += (size_t)kH * kS * kDk;  // (h,s,d)
  unsigned short* obf = w; w += (size_t)kB * kS * kD;   // (b, s, h*d)

  dim3 cb(256), cg((unsigned)((nW + 255) / 256));
  cvt_bf16_kernel<<<cg, cb, 0, stream>>>(Wq, Wqb, (int)nW);
  cvt_bf16_kernel<<<cg, cb, 0, stream>>>(Wk, Wkb, (int)nW);
  cvt_bf16_kernel<<<cg, cb, 0, stream>>>(Wv, Wvb, (int)nW);
  cvt_bf16_kernel<<<cg, cb, 0, stream>>>(Wr, Wrb, (int)nW);
  cvt_bf16_kernel<<<cg, cb, 0, stream>>>(Wo, Wob, (int)nW);

  dim3 pb(32);
  dim3 gBS(kD / 64, (kB * kS) / 32);
  proj_kernel<false><<<gBS, pb, 0, stream>>>(query, nullptr, Wqb, bq, r_w_bias,
                                             r_bias, qwb, qrb, nullptr, kB * kS, 1);
  proj_kernel<false><<<gBS, pb, 0, stream>>>(key_, nullptr, Wkb, bk, nullptr,
                                             nullptr, kbf, nullptr, nullptr,
                                             kB * kS, 0);
  proj_kernel<false><<<gBS, pb, 0, stream>>>(value, nullptr, Wvb, bv, nullptr,
                                             nullptr, vtb, nullptr, nullptr,
                                             kB * kS, 2);
  dim3 gR(kD / 64, kS / 32);
  proj_kernel<false><<<gR, pb, 0, stream>>>(r_emb, nullptr, Wrb, nullptr, nullptr,
                                            nullptr, rrb, nullptr, nullptr, kS, 3);

  int smem = 87040;  // 68KB BD + 1KB P + 16KB TDM K/V double buffer
  hipFuncSetAttribute((const void*)attn_kernel,
                      hipFuncAttributeMaxDynamicSharedMemorySize, smem);
  attn_kernel<<<dim3(kS / 16, kH, kB), pb, smem, stream>>>(qwb, qrb, kbf, vtb,
                                                           rrb, obf);

  proj_kernel<true><<<gBS, pb, 0, stream>>>(nullptr, obf, Wob, bo, nullptr,
                                            nullptr, nullptr, nullptr, out,
                                            kB * kS, 4);
}
